// EncoderViT_2448131359057
// MI455X (gfx1250) — compile-verified, matrix operand order checked by reading, not run on hardware
//
#include <hip/hip_runtime.h>
#include <hip/hip_bf16.h>
#include <math.h>

// ---------------------------------------------------------------------------
// Types for CDNA5 WMMA (wave32): 16x16x32 bf16 -> f32 accum
// ---------------------------------------------------------------------------
typedef __bf16 bf16_t;
typedef __attribute__((ext_vector_type(8)))  bf16_t v8bf;
typedef __attribute__((ext_vector_type(16))) bf16_t v16bf;
typedef __attribute__((ext_vector_type(8)))  float  v8f;

#define FLAG_BIAS  1
#define FLAG_GELU  2
#define FLAG_RES   4
#define FLAG_OBF16 8

__device__ __forceinline__ int imin(int a, int b) { return a < b ? a : b; }

// Per-lane A/B operand fetch for V_WMMA_*_16X16X32_BF16 from a row of >=32 K
// values (ISA 7.12.2): lane row = lane&15, elements 0..7 = K[kbase+0..7],
// elements 8..15 = K[kbase+16+0..7], kbase = (lane>>4)*8.
__device__ __forceinline__ v16bf load_op16(const bf16_t* rowp, int k0, int lane) {
  const int kb = k0 + ((lane >> 4) << 3);
  v8bf lo = *(const v8bf*)(rowp + kb);
  v8bf hi = *(const v8bf*)(rowp + kb + 16);
  return __builtin_shufflevector(lo, hi, 0,1,2,3,4,5,6,7,8,9,10,11,12,13,14,15);
}

__device__ __forceinline__ v8f wmma_bf16(v16bf a, v16bf b, v8f c) {
  return __builtin_amdgcn_wmma_f32_16x16x32_bf16(false, a, false, b, (short)0, c,
                                                 false, false);
}

// ---------------------------------------------------------------------------
// WMMA GEMM with double-buffered LDS staging.
//   out[M,N] = A[M,K](bf16) * WT[N,K]^T(bf16) (+bias)(+gelu)(+residual)
// Block = 256 threads = 8 waves; tile 128(M) x 64(N); K staged 64 per step
// (two 32-K WMMA sub-chunks per barrier). Global loads for step s+1 are
// issued before the WMMAs of step s; operands for each sub-chunk are loaded
// into distinct registers so the compiler can use partial DS-count waits and
// issue the four WMMAs back-to-back.
// ---------------------------------------------------------------------------
__global__ __launch_bounds__(256) void gemm_bf16_wmma(
    const bf16_t* __restrict__ A, const bf16_t* __restrict__ WT,
    const float* __restrict__ bias, const float* __restrict__ residual,
    void* __restrict__ outp, int M, int N, int K, int flags)
{
  __shared__ alignas(16) bf16_t sA[2][128 * 64];   // 2 x 16 KB
  __shared__ alignas(16) bf16_t sB[2][64 * 64];    // 2 x  8 KB

  const int tid  = threadIdx.x;
  const int lane = tid & 31;
  const int wave = tid >> 5;
  const int r16  = lane & 15;
  const int half = lane >> 4;
  const int mbase = blockIdx.y * 128;
  const int nbase = blockIdx.x * 64;

  // Staging: per step each thread moves 4 A-chunks + 2 B-chunks of 8 bf16
  // (16B each). Chunk j covers row srow + j*32, column slice part..part+7.
  const int part = (tid & 7) * 8;     // 0..56
  const int srow = tid >> 3;          // 0..31
  const bf16_t* gA[4];
  const bf16_t* gB[2];
  int ldsAo[4], ldsBo[2];
#pragma unroll
  for (int j = 0; j < 4; ++j) {
    const int row = srow + j * 32;
    gA[j]    = A + (size_t)imin(mbase + row, M - 1) * K + part;
    ldsAo[j] = row * 64 + part;
  }
#pragma unroll
  for (int j = 0; j < 2; ++j) {
    const int row = srow + j * 32;
    gB[j]    = WT + (size_t)imin(nbase + row, N - 1) * K + part;
    ldsBo[j] = row * 64 + part;
  }

  v8f zero = {};
  v8f acc[4] = { zero, zero, zero, zero };

  const int nsteps = K >> 6;

  // Prologue: stage step 0.
  v8bf rA[4], rB[2];
#pragma unroll
  for (int j = 0; j < 4; ++j) rA[j] = *(const v8bf*)gA[j];
#pragma unroll
  for (int j = 0; j < 2; ++j) rB[j] = *(const v8bf*)gB[j];
#pragma unroll
  for (int j = 0; j < 4; ++j) *(v8bf*)&sA[0][ldsAo[j]] = rA[j];
#pragma unroll
  for (int j = 0; j < 2; ++j) *(v8bf*)&sB[0][ldsBo[j]] = rB[j];
  __syncthreads();

  const bf16_t* laBase0 = &sA[0][(wave * 16 + r16) * 64];
  const bf16_t* laBase1 = &sA[1][(wave * 16 + r16) * 64];

  for (int s = 0; s < nsteps; ++s) {
    const int buf = s & 1;
    const bool more = (s + 1) < nsteps;
    if (more) {
      const int k = (s + 1) << 6;      // issued now, consumed after the WMMAs
#pragma unroll
      for (int j = 0; j < 4; ++j) rA[j] = *(const v8bf*)(gA[j] + k);
#pragma unroll
      for (int j = 0; j < 2; ++j) rB[j] = *(const v8bf*)(gB[j] + k);
    }

    const bf16_t* la = buf ? laBase1 : laBase0;
    const bf16_t* lb0 = &sB[buf][(0 * 16 + r16) * 64];
    const bf16_t* lb1 = &sB[buf][(1 * 16 + r16) * 64];
    const bf16_t* lb2 = &sB[buf][(2 * 16 + r16) * 64];
    const bf16_t* lb3 = &sB[buf][(3 * 16 + r16) * 64];

    // Sub-chunk 0 (K = 0..31 of this stage): preload all operands, then WMMA.
    {
      v16bf a  = load_op16(la,  0, lane);
      v16bf b0 = load_op16(lb0, 0, lane);
      v16bf b1 = load_op16(lb1, 0, lane);
      v16bf b2 = load_op16(lb2, 0, lane);
      v16bf b3 = load_op16(lb3, 0, lane);
      acc[0] = wmma_bf16(a, b0, acc[0]);
      acc[1] = wmma_bf16(a, b1, acc[1]);
      acc[2] = wmma_bf16(a, b2, acc[2]);
      acc[3] = wmma_bf16(a, b3, acc[3]);
    }
    // Sub-chunk 1 (K = 32..63 of this stage).
    {
      v16bf a  = load_op16(la,  32, lane);
      v16bf b0 = load_op16(lb0, 32, lane);
      v16bf b1 = load_op16(lb1, 32, lane);
      v16bf b2 = load_op16(lb2, 32, lane);
      v16bf b3 = load_op16(lb3, 32, lane);
      acc[0] = wmma_bf16(a, b0, acc[0]);
      acc[1] = wmma_bf16(a, b1, acc[1]);
      acc[2] = wmma_bf16(a, b2, acc[2]);
      acc[3] = wmma_bf16(a, b3, acc[3]);
    }

    if (more) {
      const int nb = buf ^ 1;
#pragma unroll
      for (int j = 0; j < 4; ++j) *(v8bf*)&sA[nb][ldsAo[j]] = rA[j];
#pragma unroll
      for (int j = 0; j < 2; ++j) *(v8bf*)&sB[nb][ldsBo[j]] = rB[j];
    }
    __syncthreads();
  }

  // Epilogue. C/D layout: element r of lane -> M = r + 8*(lane>>4), N = lane&15.
  const int mtile = mbase + wave * 16;
#pragma unroll
  for (int t = 0; t < 4; ++t) {
    const int col = nbase + t * 16 + r16;
    if (col >= N) continue;
    const float bv = (flags & FLAG_BIAS) ? bias[col] : 0.0f;
#pragma unroll
    for (int r = 0; r < 8; ++r) {
      const int row = mtile + r + 8 * half;
      if (row >= M) continue;
      float v = acc[t][r] + bv;
      if (flags & FLAG_GELU) v = 0.5f * v * (1.0f + erff(v * 0.70710678118f));
      if (flags & FLAG_RES)  v += residual[(size_t)row * N + col];
      if (flags & FLAG_OBF16) ((bf16_t*)outp)[(size_t)row * N + col] = (bf16_t)v;
      else                    ((float*) outp)[(size_t)row * N + col] = v;
    }
  }
}

// ---------------------------------------------------------------------------
// Attention: one wave per (batch, head, 16-query tile). S via WMMA, softmax
// over LDS-resident 197-key row, P (bf16 in LDS) x V^T via WMMA.
// ---------------------------------------------------------------------------
__global__ __launch_bounds__(32) void attn_wmma(
    const bf16_t* __restrict__ qh, const bf16_t* __restrict__ kh,
    const bf16_t* __restrict__ vT, bf16_t* __restrict__ obuf)
{
  __shared__ alignas(16) float  sm[16 * 224];
  __shared__ alignas(16) bf16_t pb[16 * 224];

  const int lane = threadIdx.x;
  const int r16  = lane & 15;
  const int half = lane >> 4;
  const int qt = blockIdx.x;   // 0..12
  const int h  = blockIdx.y;   // 0..11
  const int b  = blockIdx.z;   // 0..31

  const bf16_t* Q  = qh + ((size_t)(b * 12 + h)) * 208 * 64;
  const bf16_t* Kb = kh + ((size_t)(b * 12 + h)) * 208 * 64;
  const bf16_t* Vt = vT + ((size_t)(b * 12 + h)) * 64 * 224;
  const bf16_t* qrow = Q + (size_t)(qt * 16 + r16) * 64;

  v8f zero = {};

  // Q operands are invariant across key tiles: hoist.
  v16bf q0 = load_op16(qrow, 0,  lane);
  v16bf q1 = load_op16(qrow, 32, lane);

  // S = (Q K^T) * 1/sqrt(64); padded K rows (197..207) are zero -> S = 0.
  for (int kt = 0; kt < 13; ++kt) {
    const bf16_t* krow = Kb + (size_t)(kt * 16 + r16) * 64;
    v16bf k0v = load_op16(krow, 0,  lane);
    v16bf k1v = load_op16(krow, 32, lane);
    v8f acc = zero;
    acc = wmma_bf16(q0, k0v, acc);
    acc = wmma_bf16(q1, k1v, acc);
#pragma unroll
    for (int r = 0; r < 8; ++r)
      sm[(r + 8 * half) * 224 + kt * 16 + r16] = acc[r] * 0.125f;
  }
  __syncthreads();

  // Masked softmax over columns 0..196; lanes (l, l^16) split one row.
  {
    const int row = r16;
    const int c0 = half * 112, c1 = c0 + 112;
    float mx = -1e30f;
    for (int c = c0; c < c1; ++c)
      if (c < 197) mx = fmaxf(mx, sm[row * 224 + c]);
    mx = fmaxf(mx, __shfl_xor(mx, 16, 32));
    float sum = 0.0f;
    for (int c = c0; c < c1; ++c)
      if (c < 197) sum += __expf(sm[row * 224 + c] - mx);
    sum += __shfl_xor(sum, 16, 32);
    const float inv = 1.0f / sum;
    for (int c = c0; c < c1; ++c) {
      float p = (c < 197) ? __expf(sm[row * 224 + c] - mx) * inv : 0.0f;
      pb[row * 224 + c] = (bf16_t)p;
    }
  }
  __syncthreads();

  // O = P V : A operand streamed from LDS (bf16), B operand = rows of V^T.
#pragma unroll
  for (int dt = 0; dt < 4; ++dt) {
    v8f acc = zero;
    const bf16_t* vrow = Vt + (size_t)(dt * 16 + r16) * 224;
    for (int k0 = 0; k0 < 224; k0 += 32) {
      const int kb = k0 + (half << 3);
      v8bf lo = *(const v8bf*)&pb[r16 * 224 + kb];
      v8bf hi = *(const v8bf*)&pb[r16 * 224 + kb + 16];
      v16bf a = __builtin_shufflevector(lo, hi,
                  0,1,2,3,4,5,6,7,8,9,10,11,12,13,14,15);
      v16bf bb = load_op16(vrow, k0, lane);
      acc = wmma_bf16(a, bb, acc);
    }
#pragma unroll
    for (int r = 0; r < 8; ++r) {
      const int tl = qt * 16 + r + 8 * half;
      if (tl < 197)
        obuf[((size_t)(b * 197 + tl)) * 768 + h * 64 + dt * 16 + r16] =
            (bf16_t)acc[r];
    }
  }
}

// ---------------------------------------------------------------------------
// LayerNorm over 768 features (fp32 in -> bf16 out), one block per row.
// ---------------------------------------------------------------------------
__global__ __launch_bounds__(256) void ln_kernel(
    const float* __restrict__ in, bf16_t* __restrict__ out,
    const float* __restrict__ g, const float* __restrict__ bta,
    long long instride)
{
  const int row = blockIdx.x;
  const int t = threadIdx.x;
  const float* p = in + (size_t)row * instride;
  float x0 = p[t], x1 = p[t + 256], x2 = p[t + 512];
  float s  = x0 + x1 + x2;
  float s2 = x0 * x0 + x1 * x1 + x2 * x2;
  for (int off = 16; off > 0; off >>= 1) {
    s  += __shfl_down(s,  off, 32);
    s2 += __shfl_down(s2, off, 32);
  }
  __shared__ float ws0[8], ws1[8];
  __shared__ float mv[2];
  const int lane = t & 31, wave = t >> 5;
  if (lane == 0) { ws0[wave] = s; ws1[wave] = s2; }
  __syncthreads();
  if (t == 0) {
    float ts = 0.0f, ts2 = 0.0f;
    for (int i = 0; i < 8; ++i) { ts += ws0[i]; ts2 += ws1[i]; }
    float mean = ts * (1.0f / 768.0f);
    float var  = ts2 * (1.0f / 768.0f) - mean * mean;
    mv[0] = mean;
    mv[1] = rsqrtf(var + 1e-5f);
  }
  __syncthreads();
  const float mean = mv[0], rs = mv[1];
  bf16_t* orow = out + (size_t)row * 768;
  orow[t]       = (bf16_t)((x0 - mean) * rs * g[t]       + bta[t]);
  orow[t + 256] = (bf16_t)((x1 - mean) * rs * g[t + 256] + bta[t + 256]);
  orow[t + 512] = (bf16_t)((x2 - mean) * rs * g[t + 512] + bta[t + 512]);
}

// ---------------------------------------------------------------------------
// Elementwise helpers
// ---------------------------------------------------------------------------
__global__ void transpose_cast(const float* __restrict__ W, bf16_t* __restrict__ WT,
                               int L, int K, int N)
{
  size_t idx = (size_t)blockIdx.x * 256 + threadIdx.x;
  const size_t tot = (size_t)L * K * N;
  if (idx >= tot) return;
  const int n = (int)(idx % N);
  size_t t = idx / N;
  const int k = (int)(t % K);
  const int l = (int)(t / K);
  WT[((size_t)l * N + n) * (size_t)K + k] = (bf16_t)W[idx];
}

__global__ void im2col_patches(const float* __restrict__ img, bf16_t* __restrict__ Apat)
{
  size_t idx = (size_t)blockIdx.x * 256 + threadIdx.x;
  if (idx >= (size_t)6272 * 768) return;
  const int d = (int)(idx % 768);
  const int m = (int)(idx / 768);
  const int b = m / 196, p = m % 196;
  const int hh = p / 14, ww = p % 14;
  const int c = d % 3;
  const int pix = d / 3;
  const int py = pix / 16, px = pix % 16;
  const float v = img[(((size_t)b * 3 + c) * 224 + hh * 16 + py) * 224 + ww * 16 + px];
  Apat[idx] = (bf16_t)v;
}

__global__ void assemble_x(const float* __restrict__ xe, const float* __restrict__ cls,
                           const float* __restrict__ pos, float* __restrict__ x)
{
  size_t idx = (size_t)blockIdx.x * 256 + threadIdx.x;
  if (idx >= (size_t)6304 * 768) return;
  const int d = (int)(idx % 768);
  const int row = (int)(idx / 768);
  const int b = row / 197, t = row % 197;
  float v = pos[(size_t)t * 768 + d];
  v += (t == 0) ? cls[d] : xe[((size_t)b * 196 + (t - 1)) * 768 + d];
  x[idx] = v;
}

// qkv fp32 [6304,2304] -> q,k bf16 [B,12,208,64] (zero-padded tokens),
//                         v^T bf16 [B,12,64,224] (zero-padded keys).
__global__ void qkv_split(const float* __restrict__ qkv, bf16_t* __restrict__ qh,
                          bf16_t* __restrict__ kh, bf16_t* __restrict__ vT)
{
  size_t idx = (size_t)blockIdx.x * 256 + threadIdx.x;
  const size_t tot = (size_t)32 * 12 * 224 * 64;
  if (idx >= tot) return;
  const int d = (int)(idx & 63);
  size_t t = idx >> 6;
  const int tok = (int)(t % 224);
  size_t t2 = t / 224;
  const int h = (int)(t2 % 12);
  const int b = (int)(t2 / 12);
  float q = 0.0f, k = 0.0f, v = 0.0f;
  if (tok < 197) {
    const float* rp = qkv + ((size_t)(b * 197 + tok)) * 2304;
    q = rp[h * 64 + d];
    k = rp[768 + h * 64 + d];
    v = rp[1536 + h * 64 + d];
  }
  if (tok < 208) {
    const size_t o = (((size_t)(b * 12 + h)) * 208 + tok) * 64 + d;
    qh[o] = (bf16_t)q;
    kh[o] = (bf16_t)k;
  }
  vT[(((size_t)(b * 12 + h)) * 64 + d) * 224 + tok] = (bf16_t)v;
}

// ---------------------------------------------------------------------------
// Host orchestration
// ---------------------------------------------------------------------------
extern "C" void kernel_launch(void* const* d_in, const int* in_sizes, int n_in,
                              void* d_out, int out_size, void* d_ws, size_t ws_size,
                              hipStream_t stream)
{
  (void)in_sizes; (void)n_in; (void)out_size; (void)ws_size;

  const float* img     = (const float*)d_in[0];
  const float* patch_w = (const float*)d_in[1];
  const float* patch_b = (const float*)d_in[2];
  const float* pos_emb = (const float*)d_in[3];
  const float* cls_tok = (const float*)d_in[4];
  const float* ln1_g   = (const float*)d_in[5];
  const float* ln1_b   = (const float*)d_in[6];
  const float* qkv_w   = (const float*)d_in[7];
  const float* out_w   = (const float*)d_in[8];
  const float* out_b   = (const float*)d_in[9];
  const float* ln2_g   = (const float*)d_in[10];
  const float* ln2_b   = (const float*)d_in[11];
  const float* ff1_w   = (const float*)d_in[12];
  const float* ff1_b   = (const float*)d_in[13];
  const float* ff2_w   = (const float*)d_in[14];
  const float* ff2_b   = (const float*)d_in[15];
  const float* hln_g   = (const float*)d_in[16];
  const float* hln_b   = (const float*)d_in[17];
  const float* head_w  = (const float*)d_in[18];
  const float* head_b  = (const float*)d_in[19];

  char* ws = (char*)d_ws;
  size_t off = 0;
  auto alloc = [&](size_t bytes) -> char* {
    char* p = ws + off;
    off = (off + bytes + 255) & ~(size_t)255;
    return p;
  };

  // bf16 transposed weights
  bf16_t* patchT = (bf16_t*)alloc((size_t)768 * 768 * 2);
  bf16_t* qkvT   = (bf16_t*)alloc((size_t)12 * 2304 * 768 * 2);
  bf16_t* outT   = (bf16_t*)alloc((size_t)12 * 768 * 768 * 2);
  bf16_t* ff1T   = (bf16_t*)alloc((size_t)12 * 3072 * 768 * 2);
  bf16_t* ff2T   = (bf16_t*)alloc((size_t)12 * 768 * 3072 * 2);
  bf16_t* headT  = (bf16_t*)alloc((size_t)1000 * 768 * 2);
  // activations
  bf16_t* Apat   = (bf16_t*)alloc((size_t)6272 * 768 * 2);
  float*  xe     = (float*) alloc((size_t)6272 * 768 * 4);
  float*  x      = (float*) alloc((size_t)6304 * 768 * 4);
  bf16_t* hbuf   = (bf16_t*)alloc((size_t)6304 * 768 * 2);
  float*  qkvbuf = (float*) alloc((size_t)6304 * 2304 * 4);
  bf16_t* qh     = (bf16_t*)alloc((size_t)32 * 12 * 208 * 64 * 2);
  bf16_t* kh     = (bf16_t*)alloc((size_t)32 * 12 * 208 * 64 * 2);
  bf16_t* vTb    = (bf16_t*)alloc((size_t)32 * 12 * 64 * 224 * 2);
  bf16_t* obuf   = (bf16_t*)alloc((size_t)6304 * 768 * 2);
  bf16_t* ffbuf  = (bf16_t*)alloc((size_t)6304 * 3072 * 2);
  bf16_t* clsbuf = (bf16_t*)alloc((size_t)32 * 768 * 2);

  auto tc = [&](const float* W, bf16_t* WT, int L, int K, int N) {
    const size_t tot = (size_t)L * K * N;
    transpose_cast<<<dim3((unsigned)((tot + 255) / 256)), dim3(256), 0, stream>>>(
        W, WT, L, K, N);
  };
  auto gemm = [&](const bf16_t* A, const bf16_t* WT, const float* bias,
                  const float* res, void* out, int M, int N, int K, int flags) {
    dim3 grid((N + 63) / 64, (M + 127) / 128);
    gemm_bf16_wmma<<<grid, dim3(256), 0, stream>>>(A, WT, bias, res, out, M, N, K,
                                                   flags);
  };

  // Weight conversion (transposed to [N,K] bf16)
  tc(patch_w, patchT, 1, 768, 768);
  tc(qkv_w,   qkvT,  12, 768, 2304);
  tc(out_w,   outT,  12, 768, 768);
  tc(ff1_w,   ff1T,  12, 768, 3072);
  tc(ff2_w,   ff2T,  12, 3072, 768);
  tc(head_w,  headT,  1, 768, 1000);

  // Patch embedding + token assembly
  im2col_patches<<<dim3((unsigned)(((size_t)6272 * 768 + 255) / 256)), 256, 0,
                   stream>>>(img, Apat);
  gemm(Apat, patchT, patch_b, nullptr, xe, 6272, 768, 768, FLAG_BIAS);
  assemble_x<<<dim3((unsigned)(((size_t)6304 * 768 + 255) / 256)), 256, 0,
               stream>>>(xe, cls_tok, pos_emb, x);

  // Transformer layers
  for (int l = 0; l < 12; ++l) {
    ln_kernel<<<6304, 256, 0, stream>>>(x, hbuf, ln1_g + l * 768, ln1_b + l * 768,
                                        768);
    gemm(hbuf, qkvT + (size_t)l * 2304 * 768, nullptr, nullptr, qkvbuf,
         6304, 2304, 768, 0);
    qkv_split<<<dim3((unsigned)(((size_t)32 * 12 * 224 * 64 + 255) / 256)), 256, 0,
                stream>>>(qkvbuf, qh, kh, vTb);
    attn_wmma<<<dim3(13, 12, 32), 32, 0, stream>>>(qh, kh, vTb, obuf);
    gemm(obuf, outT + (size_t)l * 768 * 768, out_b + l * 768, x, x,
         6304, 768, 768, FLAG_BIAS | FLAG_RES);
    ln_kernel<<<6304, 256, 0, stream>>>(x, hbuf, ln2_g + l * 768, ln2_b + l * 768,
                                        768);
    gemm(hbuf, ff1T + (size_t)l * 3072 * 768, ff1_b + l * 3072, nullptr, ffbuf,
         6304, 3072, 768, FLAG_BIAS | FLAG_GELU | FLAG_OBF16);
    gemm(ffbuf, ff2T + (size_t)l * 768 * 3072, ff2_b + l * 768, x, x,
         6304, 768, 3072, FLAG_BIAS | FLAG_RES);
  }

  // Head: CLS-token LayerNorm + 32x1000 classifier
  ln_kernel<<<32, 256, 0, stream>>>(x, clsbuf, hln_g, hln_b,
                                    (long long)197 * 768);
  gemm(clsbuf, headT, head_b, nullptr, (float*)d_out, 32, 1000, 768, FLAG_BIAS);
}
